// HyperedgeConstruction_38044820308167
// MI455X (gfx1250) — matrix operand order repaired and checked by reading, not compiled, not run on hardware
//
#include <hip/hip_runtime.h>
#include <hip/hip_fp16.h>

typedef __attribute__((ext_vector_type(16))) _Float16 v16h;
typedef __attribute__((ext_vector_type(8)))  float    v8f;

#define B_N   1024
#define EMB   128
#define K2    10
#define CHUNK 256      // j-rows staged per LDS buffer
#define RP    132      // padded row stride in floats (132*4 B: 2-way bank conflict max)

// dynamic LDS layout (bytes):
//   buf0   CHUNK*RP*4      @ 0        (135168)
//   buf1   CHUNK*RP*4      @ 135168   (135168)
//   hi     EMB*4           @ 270336   (512)
//   sv     256*4           @ 270848
//   sj     256*4           @ 271872
//   swin   4               @ 272896
#define SMEM_BYTES (2 * CHUNK * RP * 4 + EMB * 4 + 256 * 4 + 256 * 4 + 4)

// ---------------------------------------------------------------------------
// ws layout (bytes):
//   h      f32 [1024*128]    @ 0         (512 KB)
//   Mt     f16 [1024*1024]   @ 524288    (2 MB)    Mt[c][i] = 1 iff c in S_i
//   A2     f32 [1024*1024]   @ 2621440   (4 MB)
//   idx    i32 [1024*10]     @ 6815744   (40 KB)
//   colcnt i32 [1024]        @ 6856704   (4 KB)
// ---------------------------------------------------------------------------

__global__ void k_prep(const float* __restrict__ t, const float* __restrict__ a,
                       const float* __restrict__ v, float* __restrict__ h,
                       float* __restrict__ out_nodes) {
  int g = blockIdx.x * blockDim.x + threadIdx.x;   // 0 .. 131071
  float tv = t[g], av = a[g], vv = v[g];
  h[g] = (tv + av + vv) * (1.0f / 3.0f);
  out_nodes[g] = tv;
  out_nodes[g + B_N * EMB] = av;
  out_nodes[g + 2 * B_N * EMB] = vv;
}

// One block per row i. j-chunks of 256 rows are double-buffered into LDS with
// global_load_async_to_lds_b128 (coalesced, ASYNCcnt-tracked), distances read
// back via ds_load_b128, then iterative block-argmax top-10 (tie -> smaller
// index, matching stable top_k).
__global__ void __launch_bounds__(256) k_dist_topk(const float* __restrict__ h,
                                                   int* __restrict__ idx_out) {
  extern __shared__ char smem[];
  float* buf0 = (float*)smem;
  float* buf1 = (float*)(smem + CHUNK * RP * 4);
  float* hi   = (float*)(smem + 2 * CHUNK * RP * 4);
  float* sv   = hi + EMB;
  int*   sj   = (int*)(sv + 256);
  int*   swin = sj + 256;

  const int i = blockIdx.x;
  const int t = threadIdx.x;

  // stage h[i] (tiny, plain loads)
  if (t < EMB / 4) ((float4*)hi)[t] = ((const float4*)h)[i * (EMB / 4) + t];

  auto issue = [&](int c) {
    // chunk c: global rows [c*CHUNK, (c+1)*CHUNK) -> buf[c&1], 16 B per lane-op
    const float4* gsrc = (const float4*)h + c * CHUNK * (EMB / 4);
    unsigned ldsbase = (unsigned)(size_t)((c & 1) ? buf1 : buf0);
#pragma unroll
    for (int r = 0; r < (CHUNK * EMB / 4) / 256; ++r) {   // 32 wave-level ops
      int m = r * 256 + t;                                // 0..8191 float4s
      int row = m >> 5, col4 = m & 31;
      unsigned loff = ldsbase + (unsigned)(row * RP * 4 + col4 * 16);
      unsigned long long gp = (unsigned long long)(size_t)(gsrc + m);
      asm volatile("global_load_async_to_lds_b128 %0, %1, off"
                   :: "v"(loff), "v"(gp) : "memory");
    }
  };

  issue(0);
  float d[4];
  for (int c = 0; c < 4; ++c) {
    if (c + 1 < 4) {
      issue(c + 1);                                   // overlaps compute of c
      asm volatile("s_wait_asynccnt %0" :: "n"(32) : "memory");  // chunk c done
    } else {
      asm volatile("s_wait_asynccnt %0" :: "n"(0) : "memory");
    }
    __syncthreads();                                  // all waves' chunk c in LDS
    const float4* rowp = (const float4*)(((c & 1) ? buf1 : buf0) + t * RP);
    const float4* hi4 = (const float4*)hi;
    float acc = 0.0f;
#pragma unroll 8
    for (int k = 0; k < EMB / 4; ++k) {
      float4 x = rowp[k];
      float4 y = hi4[k];
      acc += fabsf(x.x - y.x) + fabsf(x.y - y.y) +
             fabsf(x.z - y.z) + fabsf(x.w - y.w);
    }
    d[c] = acc;                                       // j = c*256 + t
    __syncthreads();                                  // buf reusable for chunk c+2
  }

  for (int r = 0; r < K2; ++r) {
    float bv = -1.0f;  // distances are >= 0; removed slots are -2
    int   bj = 0;
    for (int c = 0; c < 4; ++c) {                     // ascending j keeps ties stable
      if (d[c] > bv) { bv = d[c]; bj = c * 256 + t; }
    }
    sv[t] = bv; sj[t] = bj;
    __syncthreads();
    for (int off = 128; off > 0; off >>= 1) {
      if (t < off) {
        float ov = sv[t + off]; int oj = sj[t + off];
        if (ov > sv[t] || (ov == sv[t] && oj < sj[t])) { sv[t] = ov; sj[t] = oj; }
      }
      __syncthreads();
    }
    if (t == 0) { *swin = sj[0]; idx_out[i * K2 + r] = sj[0]; }
    __syncthreads();
    const int win = *swin;
    if ((win & 255) == t) d[win >> 8] = -2.0f;        // remove winner
    __syncthreads();
  }
}

// Scatter the 0/1 membership matrix Mt[c][i] and the column degrees.
__global__ void k_build(const int* __restrict__ idx, _Float16* __restrict__ Mt,
                        int* __restrict__ colcnt) {
  const int i = blockIdx.x * blockDim.x + threadIdx.x;
  if (i >= B_N) return;
  Mt[i * B_N + i] = (_Float16)1.0f;
  atomicAdd(&colcnt[i], 1);
  for (int q = 0; q < K2; ++q) {
    const int c = idx[i * K2 + q];
    Mt[c * B_N + i] = (_Float16)1.0f;       // idempotent -> binarization for free
    if (c != i) atomicAdd(&colcnt[c], 1);
  }
}

// G = Mt * Mt^T via v_wmma_f32_16x16x32_f16 (exact: 0/1 operands, int accums).
// One wave per 16x16 output tile; 64x64 tiles; K loop = 32 steps of 32.
__global__ void __launch_bounds__(256) k_gemm(const _Float16* __restrict__ Mt,
                                              float* __restrict__ A2) {
  const int lane  = threadIdx.x & 31;
  const int wave  = threadIdx.x >> 5;
  const int tile  = blockIdx.x * 8 + wave;  // 0..4095
  const int tr    = tile >> 6;
  const int tc    = tile & 63;
  const int lrow  = lane & 15;
  const int khalf = lane >> 4;

  // A-frag (16x32 f16): lane holds row M=lrow, K = 8*khalf+{0..7} and 16+8*khalf+{0..7}
  const _Float16* arow = Mt + (size_t)(tr * 16 + lrow) * B_N + 8 * khalf;
  // B-frag (32x16 f16): lane holds col N=lrow, K = 8*khalf+{0..7} and 16+8*khalf+{0..7}
  const _Float16* brow = Mt + (size_t)(tc * 16 + lrow) * B_N + 8 * khalf;

  union Frag { v16h v; uint4 u[2]; };
  v8f acc = {};
  for (int k0 = 0; k0 < B_N; k0 += 32) {
    Frag A, Bf;
    A.u[0]  = *(const uint4*)(arow + k0);
    A.u[1]  = *(const uint4*)(arow + k0 + 16);
    Bf.u[0] = *(const uint4*)(brow + k0);
    Bf.u[1] = *(const uint4*)(brow + k0 + 16);
    __builtin_prefetch(arow + k0 + 32, 0, 3);
    __builtin_prefetch(brow + k0 + 32, 0, 3);
    acc = __builtin_amdgcn_wmma_f32_16x16x32_f16(
        /*neg_a=*/false, A.v, /*neg_b=*/false, Bf.v,
        /*c_mod=*/(short)0, acc, /*reuse_a=*/false, /*reuse_b=*/false);
  }

  // C/D layout: VGPR v -> row M = v + 8*(lane>=16); col N = lane&15
  const float scale = 1.0f / 33.0f;         // common weight 1/(3*11)
  float* outp = A2 + (size_t)(tr * 16 + 8 * khalf) * B_N + tc * 16 + lrow;
#pragma unroll
  for (int vv = 0; vv < 8; ++vv) outp[(size_t)vv * B_N] = acc[vv] * scale;
}

// Rare path: rows with |S_i| == 10 (i in its own top-k) use weight 1/30.
__global__ void k_corr(const int* __restrict__ idx, float* __restrict__ A2) {
  const int i = blockIdx.x * blockDim.x + threadIdx.x;
  if (i >= B_N) return;
  int S[K2];
  bool self = false;
  for (int q = 0; q < K2; ++q) { S[q] = idx[i * K2 + q]; self |= (S[q] == i); }
  if (!self) return;                        // |S_i| == 11: GEMM scale already right
  const float delta = (1.0f / 30.0f) - (1.0f / 33.0f);
  for (int a = 0; a < K2; ++a)
    for (int b = 0; b < K2; ++b)
      atomicAdd(&A2[(size_t)S[a] * B_N + S[b]], delta);
}

// out[n1,n2] = (1{c1==c2}/3 + A2[c1,c2]) / (1 + colcnt[c1]), tiled 3x3.
__global__ void k_final(const float* __restrict__ A2, const int* __restrict__ colcnt,
                        float* __restrict__ out) {
  const int n1 = blockIdx.y;                             // 0..3071
  const int c1 = n1 & (B_N - 1);
  const int n2 = (blockIdx.x * blockDim.x + threadIdx.x) * 4;  // 0..3068
  const int c2 = n2 & (B_N - 1);
  const float inv = 1.0f / (1.0f + (float)colcnt[c1]);
  float4 a = *(const float4*)(A2 + (size_t)c1 * B_N + c2);
  float4 r;
  r.x = (a.x + ((c2 + 0) == c1 ? (1.0f / 3.0f) : 0.0f)) * inv;
  r.y = (a.y + ((c2 + 1) == c1 ? (1.0f / 3.0f) : 0.0f)) * inv;
  r.z = (a.z + ((c2 + 2) == c1 ? (1.0f / 3.0f) : 0.0f)) * inv;
  r.w = (a.w + ((c2 + 3) == c1 ? (1.0f / 3.0f) : 0.0f)) * inv;
  *(float4*)(out + (size_t)n1 * (3 * B_N) + n2) = r;
}

extern "C" void kernel_launch(void* const* d_in, const int* in_sizes, int n_in,
                              void* d_out, int out_size, void* d_ws, size_t ws_size,
                              hipStream_t stream) {
  (void)in_sizes; (void)n_in; (void)out_size; (void)ws_size;
  const float* t = (const float*)d_in[0];
  const float* a = (const float*)d_in[1];
  const float* v = (const float*)d_in[2];
  float* out = (float*)d_out;
  char* ws = (char*)d_ws;

  float*    h      = (float*)(ws + 0);
  _Float16* Mt     = (_Float16*)(ws + 524288);
  float*    A2     = (float*)(ws + 2621440);
  int*      idx    = (int*)(ws + 6815744);
  int*      colcnt = (int*)(ws + 6856704);
  float*    out_nodes = out + (size_t)(3 * B_N) * (3 * B_N);

  hipMemsetAsync(Mt, 0, (size_t)B_N * B_N * sizeof(_Float16), stream);
  hipMemsetAsync(colcnt, 0, B_N * sizeof(int), stream);

  k_prep<<<dim3((B_N * EMB) / 256), dim3(256), 0, stream>>>(t, a, v, h, out_nodes);
  k_dist_topk<<<dim3(B_N), dim3(256), SMEM_BYTES, stream>>>(h, idx);
  k_build<<<dim3(4), dim3(256), 0, stream>>>(idx, Mt, colcnt);
  k_gemm<<<dim3(512), dim3(256), 0, stream>>>(Mt, A2);
  k_corr<<<dim3(4), dim3(256), 0, stream>>>(idx, A2);
  k_final<<<dim3(3, 3 * B_N), dim3(256), 0, stream>>>(A2, colcnt, out);
}
